// F_5437428597176
// MI455X (gfx1250) — compile-verified, matrix operand order checked by reading, not run on hardware
//
#include <hip/hip_runtime.h>
#include <hip/hip_bf16.h>
#include <math.h>

// Problem constants (from the reference)
#define GB   64            // batch
#define GN   307           // nodes
#define GH   64            // hidden (K of the GEMM)
#define GD   768           // out channels = H*C
#define GE   2456          // random edges
#define GET  (GE + GN)     // edges + self loops
#define MROWS (GB * GN)    // 19648 GEMM rows
#define NEG_SLOPE 0.2f

typedef __attribute__((ext_vector_type(16))) _Float16 v16h;
typedef __attribute__((ext_vector_type(8)))  _Float16 v8h;
typedef __attribute__((ext_vector_type(8)))  float    v8f;

// ---------- order-preserving float <-> u32 encoding for atomic max ----------
__device__ __forceinline__ unsigned enc_f32(float f) {
    unsigned u = __float_as_uint(f);
    return (u & 0x80000000u) ? ~u : (u | 0x80000000u);
}
__device__ __forceinline__ float dec_f32(unsigned u) {
    unsigned b = (u & 0x80000000u) ? (u & 0x7fffffffu) : ~u;
    return __uint_as_float(b);
}

// ---------------------------------------------------------------------------
// K0: init per-node softmax accumulators + f16 conversion / transposition.
//  - zh[row][k]  = (f16) z[row][k]                (coalesced)
//  - WlT[d][h]   = (f16) Wl[h][d], same for Wr    (96 KB each -> L2 resident)
// ---------------------------------------------------------------------------
__global__ __launch_bounds__(256) void k0_prep(
    const float* __restrict__ z,
    const float* __restrict__ Wl, const float* __restrict__ Wr,
    _Float16* __restrict__ zh,
    _Float16* __restrict__ WlT, _Float16* __restrict__ WrT,
    unsigned* __restrict__ nmax, float* __restrict__ denom)
{
    int i = blockIdx.x * blockDim.x + threadIdx.x;
    if (i < MROWS * GH) zh[i] = (_Float16)z[i];
    if (i < GH * GD) {
        int h = i / GD, d = i - h * GD;
        WlT[d * GH + h] = (_Float16)Wl[i];
        WrT[d * GH + h] = (_Float16)Wr[i];
    }
    if (i < GN * GB) { nmax[i] = 0u; denom[i] = 0.0f; }
}

// Build a 16-half WMMA fragment from two contiguous 8-half runs.
// ISA 7.12.2: halves 0..7 -> K = base + 8*sel + i ; halves 8..15 -> +16.
__device__ __forceinline__ v16h load_frag(const _Float16* __restrict__ p,
                                          int k0, int sel) {
    v8h lo = *(const v8h*)(p + k0 + 8 * sel);
    v8h hi = *(const v8h*)(p + k0 + 16 + 8 * sel);
    return __builtin_shufflevector(lo, hi, 0, 1, 2, 3, 4, 5, 6, 7,
                                           8, 9, 10, 11, 12, 13, 14, 15);
}

// ---------------------------------------------------------------------------
// K1: fused dual projection xl = z@Wl + bl, xr = z@Wr + br via WMMA f16->f32.
// One wave per 16x16 tile of BOTH outputs; A fragment reused for Wl and Wr.
// All fragment loads are b128 (two per fragment per K-step).
// ---------------------------------------------------------------------------
__global__ __launch_bounds__(256) void k1_proj_wmma(
    const _Float16* __restrict__ zh,   // [MROWS, GH]
    const _Float16* __restrict__ WlT,  // [GD, GH]
    const _Float16* __restrict__ WrT,  // [GD, GH]
    const float* __restrict__ bl, const float* __restrict__ br,
    float* __restrict__ xl, float* __restrict__ xr)
{
    const int ntilesN = GD / 16;                          // 48
    int wave = (blockIdx.x * blockDim.x + threadIdx.x) >> 5;
    int lane = threadIdx.x & 31;
    int tm = wave / ntilesN;
    int tn = wave % ntilesN;
    if (tm * 16 >= MROWS) return;                         // wave-uniform guard

    int m   = lane & 15;
    int sel = lane >> 4;
    const _Float16* pa = zh  + (size_t)(tm * 16 + m) * GH;  // A row
    const _Float16* pl = WlT + (size_t)(tn * 16 + m) * GH;  // B col (transposed)
    const _Float16* pr = WrT + (size_t)(tn * 16 + m) * GH;

    v8f accL = {};
    v8f accR = {};
#pragma unroll
    for (int k0 = 0; k0 < GH; k0 += 32) {
        v16h a  = load_frag(pa, k0, sel);
        v16h bL = load_frag(pl, k0, sel);
        v16h bR = load_frag(pr, k0, sel);
        accL = __builtin_amdgcn_wmma_f32_16x16x32_f16(
            false, a, false, bL, (short)0, accL, false, false);
        accR = __builtin_amdgcn_wmma_f32_16x16x32_f16(
            false, a, false, bR, (short)0, accR, false, false);
    }

    // C/D layout: VGPR r -> (M = r + 8*sel, N = lane&15)
    int ocol = tn * 16 + (lane & 15);
    float vbl = bl[ocol];
    float vbr = br[ocol];
#pragma unroll
    for (int r = 0; r < 8; ++r) {
        size_t orow = (size_t)(tm * 16 + r + 8 * sel);
        xl[orow * GD + ocol] = accL[r] + vbl;
        xr[orow * GD + ocol] = accR[r] + vbr;
    }
}

// ---------------------------------------------------------------------------
// K2: per (edge, batch) logit  e = att . leaky_relu(xl[src]+xr[dst])
// One wave per (e,b); float4 loads (6 iters over D=768); wave reduce;
// lane0 stores logit + atomic segment max (u32 order-encoded).
// ---------------------------------------------------------------------------
__global__ __launch_bounds__(256) void k2_edge_logits(
    const float* __restrict__ xl, const float* __restrict__ xr,
    const int* __restrict__ r1, const int* __restrict__ r2,
    const float* __restrict__ att,
    float* __restrict__ esc, unsigned* __restrict__ nmax)
{
    int gw = (blockIdx.x * blockDim.x + threadIdx.x) >> 5;
    int lane = threadIdx.x & 31;
    if (gw >= GET * GB) return;
    int e = gw / GB;
    int b = gw - e * GB;
    int s = (e < GE) ? r1[e] : (e - GE);
    int d = (e < GE) ? r2[e] : (e - GE);

    const float4* pl = (const float4*)(xl + (size_t)(b * GN + s) * GD);
    const float4* pr = (const float4*)(xr + (size_t)(b * GN + d) * GD);
    const float4* pa = (const float4*)att;

    float acc = 0.0f;
#pragma unroll
    for (int j = lane; j < GD / 4; j += 32) {
        float4 u = pl[j], v = pr[j], w = pa[j];
        float h0 = u.x + v.x, h1 = u.y + v.y, h2 = u.z + v.z, h3 = u.w + v.w;
        h0 = (h0 > 0.0f) ? h0 : NEG_SLOPE * h0;
        h1 = (h1 > 0.0f) ? h1 : NEG_SLOPE * h1;
        h2 = (h2 > 0.0f) ? h2 : NEG_SLOPE * h2;
        h3 = (h3 > 0.0f) ? h3 : NEG_SLOPE * h3;
        acc += h0 * w.x + h1 * w.y + h2 * w.z + h3 * w.w;
    }
#pragma unroll
    for (int off = 16; off > 0; off >>= 1)
        acc += __shfl_xor(acc, off, 32);
    if (lane == 0) {
        esc[(size_t)e * GB + b] = acc;
        atomicMax(&nmax[d * GB + b], enc_f32(acc));
    }
}

// ---------------------------------------------------------------------------
// K3: ee = exp(e - segmax[dst]);  denom[dst] += ee   (one thread per (e,b))
// ---------------------------------------------------------------------------
__global__ __launch_bounds__(256) void k3_expsum(
    const float* __restrict__ esc,
    const int* __restrict__ r2,
    const unsigned* __restrict__ nmax,
    float* __restrict__ eexp, float* __restrict__ denom)
{
    int i = blockIdx.x * blockDim.x + threadIdx.x;
    if (i >= GET * GB) return;
    int e = i / GB;
    int b = i - e * GB;
    int d = (e < GE) ? r2[e] : (e - GE);
    float ee = __expf(esc[i] - dec_f32(nmax[d * GB + b]));
    eexp[i] = ee;
    atomicAdd(&denom[d * GB + b], ee);
}

// ---------------------------------------------------------------------------
// K4: out[row, :] = bias  (float4 stores; rows ordered b*N + n as in ref)
// ---------------------------------------------------------------------------
__global__ __launch_bounds__(256) void k4_bias(const float* __restrict__ bias,
                                               float* __restrict__ out) {
    size_t i = (size_t)blockIdx.x * blockDim.x + threadIdx.x;
    if (i < (size_t)MROWS * (GD / 4)) {
        ((float4*)out)[i] = ((const float4*)bias)[i % (GD / 4)];
    }
}

// ---------------------------------------------------------------------------
// K5: scatter  out[dst,b,:] += (ee/denom[dst,b]) * xl[src,b,:]
// One wave per (e,b); float4 loads, f32 atomic adds (order-insensitive sum).
// ---------------------------------------------------------------------------
__global__ __launch_bounds__(256) void k5_aggregate(
    const float* __restrict__ xl,
    const int* __restrict__ r1, const int* __restrict__ r2,
    const float* __restrict__ eexp, const float* __restrict__ denom,
    float* __restrict__ out)
{
    int gw = (blockIdx.x * blockDim.x + threadIdx.x) >> 5;
    int lane = threadIdx.x & 31;
    if (gw >= GET * GB) return;
    int e = gw / GB;
    int b = gw - e * GB;
    int s = (e < GE) ? r1[e] : (e - GE);
    int d = (e < GE) ? r2[e] : (e - GE);

    float alpha = eexp[(size_t)e * GB + b] / denom[d * GB + b];
    const float4* pl = (const float4*)(xl + (size_t)(b * GN + s) * GD);
    float* po = out + (size_t)(b * GN + d) * GD;
#pragma unroll
    for (int j = lane; j < GD / 4; j += 32) {
        float4 u = pl[j];
        atomicAdd(&po[4 * j + 0], alpha * u.x);
        atomicAdd(&po[4 * j + 1], alpha * u.y);
        atomicAdd(&po[4 * j + 2], alpha * u.z);
        atomicAdd(&po[4 * j + 3], alpha * u.w);
    }
}

// ---------------------------------------------------------------------------
extern "C" void kernel_launch(void* const* d_in, const int* in_sizes, int n_in,
                              void* d_out, int out_size, void* d_ws, size_t ws_size,
                              hipStream_t stream) {
    // setup_inputs order: t, z, r1, r2, Wl, bl, Wr, br, att, bias
    const float* z    = (const float*)d_in[1];
    const int*   r1   = (const int*)  d_in[2];
    const int*   r2   = (const int*)  d_in[3];
    const float* Wl   = (const float*)d_in[4];
    const float* bl   = (const float*)d_in[5];
    const float* Wr   = (const float*)d_in[6];
    const float* br   = (const float*)d_in[7];
    const float* att  = (const float*)d_in[8];
    const float* bias = (const float*)d_in[9];
    float* out = (float*)d_out;

    // workspace carve-up
    float* ws = (float*)d_ws;
    size_t MD = (size_t)MROWS * GD;          // 15,089,664 floats
    float*    xl    = ws;
    float*    xr    = xl + MD;
    float*    esc   = xr + MD;
    float*    eexp  = esc  + (size_t)GET * GB;
    unsigned* nmax  = (unsigned*)(eexp + (size_t)GET * GB);
    float*    denom = (float*)(nmax + (size_t)GN * GB);
    _Float16* zh    = (_Float16*)(denom + (size_t)GN * GB);
    _Float16* WlT   = zh  + (size_t)MROWS * GH;
    _Float16* WrT   = WlT + (size_t)GD * GH;

    // K0: f16 conversion / W transpose / accumulator init
    {
        int n = MROWS * GH;                   // covers all three ranges
        k0_prep<<<(n + 255) / 256, 256, 0, stream>>>(z, Wl, Wr, zh, WlT, WrT,
                                                     nmax, denom);
    }
    // K1: WMMA projections (one wave per 16x16 tile; 1228*48 waves)
    {
        int waves = (MROWS / 16) * (GD / 16);
        int blocks = (waves * 32 + 255) / 256;
        k1_proj_wmma<<<blocks, 256, 0, stream>>>(zh, WlT, WrT, bl, br, xl, xr);
    }
    // K2: edge logits + segment max
    {
        int waves = GET * GB;                 // 176,832
        int blocks = (waves * 32 + 255) / 256;
        k2_edge_logits<<<blocks, 256, 0, stream>>>(xl, xr, r1, r2, att, esc, nmax);
    }
    // K3: exp + segment sum
    {
        int n = GET * GB;
        k3_expsum<<<(n + 255) / 256, 256, 0, stream>>>(esc, r2, nmax, eexp, denom);
    }
    // K4: out = bias
    {
        size_t n = (size_t)MROWS * (GD / 4);
        k4_bias<<<(unsigned)((n + 255) / 256), 256, 0, stream>>>(bias, out);
    }
    // K5: weighted scatter-add aggregation
    {
        int waves = GET * GB;
        int blocks = (waves * 32 + 255) / 256;
        k5_aggregate<<<blocks, 256, 0, stream>>>(xl, r1, r2, eexp, denom, out);
    }
}
